// SelfAttentionLayer_80393197846598
// MI455X (gfx1250) — matrix operand order repaired and checked by reading, not compiled, not run on hardware
//
#include <hip/hip_runtime.h>
#include <hip/hip_bf16.h>

// ---------------------------------------------------------------------------
// Self-attention forward for MI455X (gfx1250), bf16 WMMA pipeline.
// B=2, S=2048, D=1024, H=16, hd=64. Outputs: out [B,S,D] fp32 then
// attn [B,H,S,S] fp32, concatenated in d_out.
//
// Roofline: ~69 GFLOP total (trivial at WMMA rates); dominant cost is the
// 1.9 GB of HBM traffic materializing attn (write + softmax rw + ctx read).
// Design: everything except attn kept in bf16 (48 MB ws, L2-resident),
// causal mask fused into scores epilogue, fully-masked tiles skip WMMA,
// ctx K-loop causally truncated, all GEMM loops software-pipelined.
// ---------------------------------------------------------------------------

typedef __attribute__((ext_vector_type(16))) __bf16 v16bf;
typedef __attribute__((ext_vector_type(8)))  float  v8f;

#define CB 2
#define CS 2048
#define CD 1024
#define CH 16
#define CHD 64
#define NEG_INF (-100000.0f)

// ---- bf16 helpers (round-to-nearest-even) ---------------------------------
__device__ __forceinline__ unsigned short f32_to_bf16(float f) {
  unsigned int u = __builtin_bit_cast(unsigned int, f);
  u += 0x7FFFu + ((u >> 16) & 1u);
  return (unsigned short)(u >> 16);
}
__device__ __forceinline__ unsigned int pack2_bf16(float lo, float hi) {
  return (unsigned int)f32_to_bf16(lo) | ((unsigned int)f32_to_bf16(hi) << 16);
}

// ---- WMMA fragment loaders (ISA 7.12.2 layouts, wave32) -------------------
// A 16x32 bf16: lane half h=l>>4, row m=l&15; VGPRs 0-3 hold K=kbase+8h..+7,
// VGPRs 4-7 hold K=kbase+16+8h..+7  -> two 16-byte loads per lane.
__device__ __forceinline__ v16bf load_a_bf16(const unsigned short* A, int lda,
                                             int mbase, int kbase, int lane) {
  union { v16bf v; uint4 q[2]; } f;
  const unsigned short* p = A + (size_t)(mbase + (lane & 15)) * lda + kbase +
                            ((lane >> 4) << 3);
  f.q[0] = *reinterpret_cast<const uint4*>(p);
  f.q[1] = *reinterpret_cast<const uint4*>(p + 16);
  return f.v;
}
// Same A fragment but source is fp32 (attn): 4x float4 loads, pack to bf16.
__device__ __forceinline__ v16bf load_a_f32(const float* A, int lda,
                                            int mbase, int kbase, int lane) {
  union { v16bf v; unsigned int u[8]; } f;
  const float* p = A + (size_t)(mbase + (lane & 15)) * lda + kbase +
                   ((lane >> 4) << 3);
  float4 f0 = *reinterpret_cast<const float4*>(p);
  float4 f1 = *reinterpret_cast<const float4*>(p + 4);
  float4 f2 = *reinterpret_cast<const float4*>(p + 16);
  float4 f3 = *reinterpret_cast<const float4*>(p + 20);
  f.u[0] = pack2_bf16(f0.x, f0.y); f.u[1] = pack2_bf16(f0.z, f0.w);
  f.u[2] = pack2_bf16(f1.x, f1.y); f.u[3] = pack2_bf16(f1.z, f1.w);
  f.u[4] = pack2_bf16(f2.x, f2.y); f.u[5] = pack2_bf16(f2.z, f2.w);
  f.u[6] = pack2_bf16(f3.x, f3.y); f.u[7] = pack2_bf16(f3.z, f3.w);
  return f.v;
}
// B 32x16 bf16 from BT[N][K]: lane col n=l&15; elems 0-7 K=kbase+8h..+7,
// elems 8-15 K=kbase+16+8h..+7 (h=l>>4) -> two 16-byte loads.
__device__ __forceinline__ v16bf load_bt_bf16(const unsigned short* BT, int ldb,
                                              int nbase, int kbase, int lane) {
  union { v16bf v; uint4 q[2]; } f;
  const unsigned short* p = BT + (size_t)(nbase + (lane & 15)) * ldb + kbase +
                            ((lane >> 4) << 3);
  f.q[0] = *reinterpret_cast<const uint4*>(p);
  f.q[1] = *reinterpret_cast<const uint4*>(p + 16);
  return f.v;
}

__device__ __forceinline__ v8f wmma_bf16(v16bf a, v16bf b, v8f c) {
  return __builtin_amdgcn_wmma_f32_16x16x32_bf16(false, a, false, b,
                                                 (short)0, c, false, false);
}

// ---------------------------------------------------------------------------
// K1: fp32 -> bf16 cast (x), vectorized: float4 in, 4x bf16 (uint2) out.
__global__ void cast_x_kernel(const float* __restrict__ in,
                              unsigned int* __restrict__ out, int n4) {
  int i = blockIdx.x * blockDim.x + threadIdx.x;
  if (i < n4) {
    float4 f = reinterpret_cast<const float4*>(in)[i];
    out[2 * i]     = pack2_bf16(f.x, f.y);
    out[2 * i + 1] = pack2_bf16(f.z, f.w);
  }
}

// K2: W [K][N] fp32 -> Wt [N][K] bf16, LDS-tiled 32x32 transpose.
__global__ void tcast_kernel(const float* __restrict__ W,
                             unsigned short* __restrict__ Wt) {
  __shared__ unsigned short tile[32][33];
  const int tx = threadIdx.x & 31, ty = threadIdx.x >> 5;  // 32x8
  const int bx = blockIdx.x & (CD / 32 - 1);               // n-tile
  const int by = blockIdx.x / (CD / 32);                   // k-tile
#pragma unroll
  for (int i = 0; i < 4; ++i) {
    int k = by * 32 + ty + i * 8, n = bx * 32 + tx;
    tile[ty + i * 8][tx] = f32_to_bf16(W[(size_t)k * CD + n]);
  }
  __syncthreads();
#pragma unroll
  for (int i = 0; i < 4; ++i) {
    int n = bx * 32 + ty + i * 8, k = by * 32 + tx;
    Wt[(size_t)n * CD + k] = tile[tx][ty + i * 8];
  }
}

// K3: fused QKV projection. M=B*S=4096, N=D=1024, K=D=1024.
// A fragments (x) loaded once per k-step, reused for 3 WMMA chains.
// Software-pipelined: next k-step's fragments are in flight during WMMA.
// Q,K -> [B,H,S,hd] bf16 ; V -> [B,H,hd,S] bf16 (transposed for ctx GEMM).
__global__ void qkv_kernel(const unsigned short* __restrict__ xb,
                           const unsigned short* __restrict__ Wqt,
                           const unsigned short* __restrict__ Wkt,
                           const unsigned short* __restrict__ Wvt,
                           unsigned short* __restrict__ qb,
                           unsigned short* __restrict__ kb,
                           unsigned short* __restrict__ vb) {
  const int lane = threadIdx.x & 31;
  const int tile = blockIdx.x * 4 + (threadIdx.x >> 5);
  const int mbase = (tile >> 6) << 4;   // 64 n-tiles
  const int nbase = (tile & 63) << 4;
  v8f cq = {}, ck = {}, cv = {};
  v16bf a  = load_a_bf16(xb, CD, mbase, 0, lane);
  v16bf bq = load_bt_bf16(Wqt, CD, nbase, 0, lane);
  v16bf bk = load_bt_bf16(Wkt, CD, nbase, 0, lane);
  v16bf bv = load_bt_bf16(Wvt, CD, nbase, 0, lane);
  for (int k = 0; k < CD; k += 32) {
    const int kn = (k + 32) & (CD - 1);      // wraps to 0 (dummy) on last iter
    v16bf a2  = load_a_bf16(xb, CD, mbase, kn, lane);
    v16bf bq2 = load_bt_bf16(Wqt, CD, nbase, kn, lane);
    v16bf bk2 = load_bt_bf16(Wkt, CD, nbase, kn, lane);
    v16bf bv2 = load_bt_bf16(Wvt, CD, nbase, kn, lane);
    cq = wmma_bf16(a, bq, cq);
    ck = wmma_bf16(a, bk, ck);
    cv = wmma_bf16(a, bv, cv);
    a = a2; bq = bq2; bk = bk2; bv = bv2;
  }
  const int n = nbase + (lane & 15);
  const int h = n >> 6, d = n & 63;
#pragma unroll
  for (int v = 0; v < 8; ++v) {
    int m = mbase + v + ((lane >> 4) << 3);
    int b = m >> 11, s = m & (CS - 1);
    size_t bh = (size_t)(b * CH + h);
    qb[(bh * CS + s) * CHD + d] = f32_to_bf16(cq[v]);
    kb[(bh * CS + s) * CHD + d] = f32_to_bf16(ck[v]);
    vb[(bh * CHD + d) * CS + s] = f32_to_bf16(cv[v]);
  }
}

// K4: scores = Q K^T / 8, causal mask -> attn buffer (raw, pre-softmax).
// One wave per 16x16 tile; tiles strictly above diagonal skip WMMA.
__global__ void scores_kernel(const unsigned short* __restrict__ qb,
                              const unsigned short* __restrict__ kb,
                              float* __restrict__ attn) {
  const int lane = threadIdx.x & 31;
  const int tile = blockIdx.x * 4 + (threadIdx.x >> 5);
  const int bh = tile >> 14;            // 128*128 tiles per (b,h)
  const int t = tile & 16383;
  const int qt = t >> 7, kt = t & 127;
  const int mbase = qt << 4, nbase = kt << 4;
  float* out = attn + (size_t)bh * CS * CS;
  const int n = nbase + (lane & 15);
  if (kt > qt) {                        // fully masked tile
#pragma unroll
    for (int v = 0; v < 8; ++v) {
      int m = mbase + v + ((lane >> 4) << 3);
      out[(size_t)m * CS + n] = NEG_INF;
    }
    return;
  }
  const unsigned short* Q = qb + (size_t)bh * CS * CHD;
  const unsigned short* K = kb + (size_t)bh * CS * CHD;  // serves as B^T
  // Issue all loads before the first WMMA so the second overlaps.
  v16bf a0 = load_a_bf16(Q, CHD, mbase, 0, lane);
  v16bf b0 = load_bt_bf16(K, CHD, nbase, 0, lane);
  v16bf a1 = load_a_bf16(Q, CHD, mbase, 32, lane);
  v16bf b1 = load_bt_bf16(K, CHD, nbase, 32, lane);
  v8f c = {};
  c = wmma_bf16(a0, b0, c);
  c = wmma_bf16(a1, b1, c);
#pragma unroll
  for (int v = 0; v < 8; ++v) {
    int m = mbase + v + ((lane >> 4) << 3);
    float val = (n > m) ? NEG_INF : c[v] * 0.125f;
    out[(size_t)m * CS + n] = val;
  }
}

// K5: row softmax over attn (in place). One 256-thread block per row.
__global__ void softmax_kernel(float* __restrict__ attn) {
  __shared__ float red[256];
  float* row = attn + (size_t)blockIdx.x * CS;
  const int t = threadIdx.x;
  float vals[8];
  float mx = -3.4e38f;
#pragma unroll
  for (int i = 0; i < 8; ++i) { vals[i] = row[t + i * 256]; mx = fmaxf(mx, vals[i]); }
  red[t] = mx; __syncthreads();
  for (int s = 128; s > 0; s >>= 1) {
    if (t < s) red[t] = fmaxf(red[t], red[t + s]);
    __syncthreads();
  }
  mx = red[0]; __syncthreads();
  float sum = 0.f;
#pragma unroll
  for (int i = 0; i < 8; ++i) { vals[i] = __expf(vals[i] - mx); sum += vals[i]; }
  red[t] = sum; __syncthreads();
  for (int s = 128; s > 0; s >>= 1) {
    if (t < s) red[t] += red[t + s];
    __syncthreads();
  }
  float inv = 1.0f / red[0];
#pragma unroll
  for (int i = 0; i < 8; ++i) row[t + i * 256] = vals[i] * inv;
}

// K6: ctx = attn @ V. M=S, N=hd=64, K=S (causally truncated).
// One wave computes a full 16x64 strip (4 accumulators) so each fp32 attn
// fragment is read and converted exactly once. Software-pipelined.
__global__ void ctx_kernel(const float* __restrict__ attn,
                           const unsigned short* __restrict__ vb,
                           unsigned short* __restrict__ ctx) {
  const int lane = threadIdx.x & 31;
  const int tile = blockIdx.x * 4 + (threadIdx.x >> 5);  // B*H*128 tiles
  const int bh = tile >> 7;
  const int qt = tile & 127;
  const int mbase = qt << 4;
  const float* A = attn + (size_t)bh * CS * CS;
  const unsigned short* BT = vb + (size_t)bh * CHD * CS;  // [hd][S]
  v8f acc[4] = {{}, {}, {}, {}};
  const int ksteps = (qt >> 1) + 1;     // attn is zero beyond the diagonal
  v16bf a = load_a_f32(A, CS, mbase, 0, lane);
  v16bf b[4], bn[4];
#pragma unroll
  for (int nt = 0; nt < 4; ++nt) b[nt] = load_bt_bf16(BT, CS, nt << 4, 0, lane);
  for (int ks = 0; ks < ksteps; ++ks) {
    const int kn = (ks + 1 < ksteps) ? ((ks + 1) << 5) : 0;  // dummy on last
    v16bf a2 = load_a_f32(A, CS, mbase, kn, lane);
#pragma unroll
    for (int nt = 0; nt < 4; ++nt)
      bn[nt] = load_bt_bf16(BT, CS, nt << 4, kn, lane);
#pragma unroll
    for (int nt = 0; nt < 4; ++nt) acc[nt] = wmma_bf16(a, b[nt], acc[nt]);
    a = a2;
#pragma unroll
    for (int nt = 0; nt < 4; ++nt) b[nt] = bn[nt];
  }
  const int bidx = bh >> 4, h = bh & 15;
#pragma unroll
  for (int nt = 0; nt < 4; ++nt) {
    int n = (nt << 4) + (lane & 15);
#pragma unroll
    for (int v = 0; v < 8; ++v) {
      int m = mbase + v + ((lane >> 4) << 3);
      ctx[((size_t)(bidx * CS + m)) * CD + h * CHD + n] = f32_to_bf16(acc[nt][v]);
    }
  }
}

// K7: out = ctx @ Wo + bo. M=4096, N=1024, K=1024, fp32 output. Pipelined.
__global__ void out_kernel(const unsigned short* __restrict__ ctx,
                           const unsigned short* __restrict__ Wot,
                           const float* __restrict__ bo,
                           float* __restrict__ out) {
  const int lane = threadIdx.x & 31;
  const int tile = blockIdx.x * 4 + (threadIdx.x >> 5);
  const int mbase = (tile >> 6) << 4;
  const int nbase = (tile & 63) << 4;
  v8f c = {};
  v16bf a = load_a_bf16(ctx, CD, mbase, 0, lane);
  v16bf b = load_bt_bf16(Wot, CD, nbase, 0, lane);
  for (int k = 0; k < CD; k += 32) {
    const int kn = (k + 32) & (CD - 1);
    v16bf a2 = load_a_bf16(ctx, CD, mbase, kn, lane);
    v16bf b2 = load_bt_bf16(Wot, CD, nbase, kn, lane);
    c = wmma_bf16(a, b, c);
    a = a2; b = b2;
  }
  const int n = nbase + (lane & 15);
  const float bias = bo[n];
#pragma unroll
  for (int v = 0; v < 8; ++v) {
    int m = mbase + v + ((lane >> 4) << 3);
    out[(size_t)m * CD + n] = c[v] + bias;
  }
}

// ---------------------------------------------------------------------------
extern "C" void kernel_launch(void* const* d_in, const int* in_sizes, int n_in,
                              void* d_out, int out_size, void* d_ws, size_t ws_size,
                              hipStream_t stream) {
  const float* x  = (const float*)d_in[0];
  const float* Wq = (const float*)d_in[1];
  const float* Wk = (const float*)d_in[2];
  const float* Wv = (const float*)d_in[3];
  const float* Wo = (const float*)d_in[4];
  const float* bo = (const float*)d_in[5];

  float* out  = (float*)d_out;                       // [B,S,D]
  float* attn = out + (size_t)CB * CS * CD;          // [B,H,S,S]

  // Workspace layout (bf16 elements). Total 48 MB.
  unsigned short* ws = (unsigned short*)d_ws;
  unsigned short* xb   = ws;                         //  4M
  unsigned short* Wqt  = xb  + (size_t)CB * CS * CD; //  1M each
  unsigned short* Wkt  = Wqt + (size_t)CD * CD;
  unsigned short* Wvt  = Wkt + (size_t)CD * CD;
  unsigned short* Wot  = Wvt + (size_t)CD * CD;
  unsigned short* qb   = Wot + (size_t)CD * CD;      //  4M  [B,H,S,hd]
  unsigned short* kb   = qb  + (size_t)CB * CS * CD; //  4M  [B,H,S,hd]
  unsigned short* vb   = kb  + (size_t)CB * CS * CD; //  4M  [B,H,hd,S]
  unsigned short* ctxb = vb  + (size_t)CB * CS * CD; //  4M  [B,S,D]
  (void)ws_size; (void)in_sizes; (void)n_in; (void)out_size;

  const int nx4 = (CB * CS * CD) / 4;
  cast_x_kernel<<<nx4 / 256, 256, 0, stream>>>(x, (unsigned int*)xb, nx4);
  tcast_kernel<<<(CD / 32) * (CD / 32), 256, 0, stream>>>(Wq, Wqt);
  tcast_kernel<<<(CD / 32) * (CD / 32), 256, 0, stream>>>(Wk, Wkt);
  tcast_kernel<<<(CD / 32) * (CD / 32), 256, 0, stream>>>(Wv, Wvt);
  tcast_kernel<<<(CD / 32) * (CD / 32), 256, 0, stream>>>(Wo, Wot);

  // QKV: 256 m-tiles * 64 n-tiles = 16384 tiles, 4 waves/block.
  qkv_kernel<<<16384 / 4, 128, 0, stream>>>(xb, Wqt, Wkt, Wvt, qb, kb, vb);

  // Scores: B*H * 128*128 tiles = 524288 tiles.
  scores_kernel<<<524288 / 4, 128, 0, stream>>>(qb, kb, attn);

  // Softmax: one block per row (B*H*S rows).
  softmax_kernel<<<CB * CH * CS, 256, 0, stream>>>(attn);

  // ctx: B*H*128 strip-tiles (16 rows x full hd), 4 waves/block.
  ctx_kernel<<<(CB * CH * 128) / 4, 128, 0, stream>>>(attn, vb, ctxb);

  // Output projection: 16384 tiles.
  out_kernel<<<16384 / 4, 128, 0, stream>>>(ctxb, Wot, bo, out);
}
